// Split_CNLB_46145128628251
// MI455X (gfx1250) — matrix operand order repaired and checked by reading, not compiled, not run on hardware
//
#include <hip/hip_runtime.h>
#include <hip/hip_bf16.h>
#include <math.h>

typedef _Float16 f16x4  __attribute__((ext_vector_type(4)));
typedef _Float16 f16x8  __attribute__((ext_vector_type(8)));
typedef _Float16 f16x16 __attribute__((ext_vector_type(16)));
typedef float    f32x8  __attribute__((ext_vector_type(8)));

#define CI      128
#define NTOT    4096
#define NSPLIT  4
#define NSLICE  (NTOT / NSPLIT)   // 1024
#define NBLK    64
#define CHUNK1  128

union FragU { f16x16 v; f16x8 h[2]; };

// A-matrix 16x32 f16 fragment: lane<16 -> M=lane, K={0..7,16..23}; lane>=16 -> M=lane-16, K={8..15,24..31}
static __device__ __forceinline__ f16x16 frag_a(const _Float16* base, int ld, int m, int k, int lane) {
  const int l = lane & 15, hi = lane >> 4;
  const f16x8* p = (const f16x8*)(base + (size_t)(m + l) * ld + k);
  FragU f; f.h[0] = p[hi]; f.h[1] = p[hi + 2];
  return f.v;
}
// B-matrix 32x16 f16 fragment: lane<16 -> N=lane, K={0..7,16..23}; lane>=16 -> N=lane-16, K={8..15,24..31}
static __device__ __forceinline__ f16x16 frag_b(const _Float16* base, int ld, int n, int k, int lane) {
  const int l = lane & 15, hi = lane >> 4;
  const f16x8* p = (const f16x8*)(base + (size_t)(n + l) * ld + k);
  FragU f; f.h[0] = p[hi]; f.h[1] = p[hi + 2];
  return f.v;
}
static __device__ __forceinline__ f32x8 wmma16(f16x16 a, f16x16 b, f32x8 c) {
  return __builtin_amdgcn_wmma_f32_16x16x32_f16(false, a, false, b, (short)0, c, false, false);
}

// ---------------- kernel 0: convert shared weights to f16 layouts ----------------
__global__ __launch_bounds__(256) void k_wconv(
    const float* __restrict__ theta_w, const float* __restrict__ phi_w,
    const float* __restrict__ g_w, const float* __restrict__ W_w,
    _Float16* __restrict__ phiH, _Float16* __restrict__ thetaH,
    _Float16* __restrict__ gwTH, _Float16* __restrict__ WwH) {
  int idx = blockIdx.x * 256 + threadIdx.x;   // 64*256 == 128*128
  phiH[idx]   = (_Float16)phi_w[idx];
  thetaH[idx] = (_Float16)theta_w[idx];
  WwH[idx]    = (_Float16)W_w[idx];
  int d = idx >> 7, i = idx & 127;
  gwTH[(size_t)i * CI + d] = (_Float16)g_w[idx]; // transpose g_w
}

// ---------------- kernel 1: per-block gram G = X X^T and row sums s ----------------
// Software-pipelined: global loads for chunk ch+1 are issued before the WMMA
// phase on chunk ch; conversion+LDS store happens after, then one barrier.
__global__ __launch_bounds__(256) void k_gram(const float* __restrict__ x,
                                              float* __restrict__ Gp,
                                              float* __restrict__ sp) {
  __shared__ _Float16 Xc[2][CI][CHUNK1];   // 2 x 32 KB ping-pong
  __shared__ float rowsum[CI];
  const int t = threadIdx.x;
  const int wg = blockIdx.x;               // 0..NBLK*NSPLIT-1
  const int b = wg / NSPLIT, spi = wg % NSPLIT;
  const int lane = t & 31, wave = t >> 5;
  const int l = lane & 15, hi = lane >> 4;
  const int strip = wave * 16;
  const int NCH = NSLICE / CHUNK1;         // 8
  const float* xb = x + (size_t)b * CI * NTOT + (size_t)spi * NSLICE;

  if (t < CI) rowsum[t] = 0.f;
  float rs[16];
#pragma unroll
  for (int i = 0; i < 16; ++i) rs[i] = 0.f;
  const f32x8 z8 = {0.f, 0.f, 0.f, 0.f, 0.f, 0.f, 0.f, 0.f};
  f32x8 acc[8];
#pragma unroll
  for (int i = 0; i < 8; ++i) acc[i] = z8;

  // prologue: chunk 0 -> buffer 0
  {
    float4 st[16];
#pragma unroll
    for (int i = 0; i < 16; ++i) {
      int idx = t + 256 * i;
      int row = idx >> 5, c4 = idx & 31;          // row = (t>>5) + 8*i
      st[i] = ((const float4*)(xb + (size_t)row * NTOT))[c4];
    }
#pragma unroll
    for (int i = 0; i < 16; ++i) {
      int idx = t + 256 * i;
      int row = idx >> 5, c4 = idx & 31;
      rs[i] += st[i].x + st[i].y + st[i].z + st[i].w;
      f16x4 h; h.x = (_Float16)st[i].x; h.y = (_Float16)st[i].y;
      h.z = (_Float16)st[i].z; h.w = (_Float16)st[i].w;
      *(f16x4*)&Xc[0][row][c4 * 4] = h;
    }
  }
  __syncthreads();

  for (int ch = 0; ch < NCH; ++ch) {
    const int cur = ch & 1, nxt = cur ^ 1;
    float4 st[16];
    if (ch + 1 < NCH) {
#pragma unroll
      for (int i = 0; i < 16; ++i) {
        int idx = t + 256 * i;
        int row = idx >> 5, c4 = idx & 31;
        st[i] = ((const float4*)(xb + (size_t)row * NTOT + (size_t)(ch + 1) * CHUNK1))[c4];
      }
    }
    // WMMA on current buffer while the global loads above are in flight
#pragma unroll
    for (int ks = 0; ks < 4; ++ks) {
      f16x16 a = frag_a(&Xc[cur][0][0], CHUNK1, strip, ks * 32, lane);
#pragma unroll
      for (int nt = 0; nt < 8; ++nt) {
        f16x16 bb = frag_b(&Xc[cur][0][0], CHUNK1, nt * 16, ks * 32, lane);
        acc[nt] = wmma16(a, bb, acc[nt]);
      }
    }
    if (ch + 1 < NCH) {
#pragma unroll
      for (int i = 0; i < 16; ++i) {
        int idx = t + 256 * i;
        int row = idx >> 5, c4 = idx & 31;
        rs[i] += st[i].x + st[i].y + st[i].z + st[i].w;
        f16x4 h; h.x = (_Float16)st[i].x; h.y = (_Float16)st[i].y;
        h.z = (_Float16)st[i].z; h.w = (_Float16)st[i].w;
        *(f16x4*)&Xc[nxt][row][c4 * 4] = h;
      }
    }
    __syncthreads();
  }

  float* Gout = Gp + (size_t)wg * CI * CI;
#pragma unroll
  for (int nt = 0; nt < 8; ++nt)
#pragma unroll
    for (int r = 0; r < 8; ++r)
      Gout[(size_t)(strip + hi * 8 + r) * CI + nt * 16 + l] = acc[nt][r];

#pragma unroll
  for (int i = 0; i < 16; ++i)
    atomicAdd(&rowsum[(t >> 5) + 8 * i], rs[i]);
  __syncthreads();
  if (t < CI) sp[(size_t)wg * CI + t] = rowsum[t];
}

// ---------------- kernel 2: attention chain, all matmuls on WMMA ----------------
__global__ __launch_bounds__(256) void k_attn(
    const float* __restrict__ Gp, const float* __restrict__ sp,
    const _Float16* __restrict__ phiH, const _Float16* __restrict__ thetaH,
    const _Float16* __restrict__ gwTH, const _Float16* __restrict__ WwH,
    const float* __restrict__ theta_b, const float* __restrict__ phi_b,
    const float* __restrict__ g_b, const float* __restrict__ W_w,
    const float* __restrict__ W_b, const float* __restrict__ bn_gamma,
    const float* __restrict__ bn_beta, const float* __restrict__ bn_mean,
    const float* __restrict__ bn_var, const float* __restrict__ phi_w,
    const float* __restrict__ theta_w,
    _Float16* __restrict__ Ah, float* __restrict__ cv) {
  __shared__ _Float16 buf0[CI * CI];  // G (symmetric => usable as B directly), later T2^T
  __shared__ _Float16 buf1[CI * CI];  // T1, later F (softmax output)
  __shared__ float sv[CI], phs[CI], ths[CI], tb[CI], pb[CI], bsc[CI], bsh[CI], uv[CI];
  const int t = threadIdx.x, b = blockIdx.x;
  const int lane = t & 31, wave = t >> 5, l = lane & 15, hi = lane >> 4;
  const int strip = wave * 16;
  const f32x8 z8 = {0.f, 0.f, 0.f, 0.f, 0.f, 0.f, 0.f, 0.f};

  // phase 0: reduce NSPLIT partial grams / sums
  const float* gp = Gp + (size_t)b * NSPLIT * CI * CI;
  for (int idx = t; idx < CI * CI; idx += 256) {
    float s = 0.f;
    for (int k2 = 0; k2 < NSPLIT; ++k2) s += gp[(size_t)k2 * CI * CI + idx];
    buf0[idx] = (_Float16)s;
  }
  if (t < CI) {
    float s = 0.f;
    for (int k2 = 0; k2 < NSPLIT; ++k2) s += sp[((size_t)b * NSPLIT + k2) * CI + t];
    sv[t] = s;
    tb[t] = theta_b[t];
    pb[t] = phi_b[t];
    float iv = rsqrtf(bn_var[t] + 1e-5f);
    bsc[t] = bn_gamma[t] * iv;
    bsh[t] = bn_beta[t] - bn_mean[t] * bsc[t];
  }
  __syncthreads();
  // phs = Phi*s, ths = Theta*s
  if (t < CI) {
    float a = 0.f; const float* r = phi_w + (size_t)t * CI;
    for (int i = 0; i < CI; ++i) a += r[i] * sv[i];
    phs[t] = a;
  } else {
    int d = t - CI;
    float a = 0.f; const float* r = theta_w + (size_t)d * CI;
    for (int i = 0; i < CI; ++i) a += r[i] * sv[i];
    ths[d] = a;
  }
  __syncthreads();

  f32x8 acc[8];
  // mm1: T1 = Phi * G   (A = phiH global, B = buf0; G symmetric)
#pragma unroll
  for (int i = 0; i < 8; ++i) acc[i] = z8;
#pragma unroll
  for (int ks = 0; ks < 4; ++ks) {
    f16x16 a = frag_a(phiH, CI, strip, ks * 32, lane);
#pragma unroll
    for (int nt = 0; nt < 8; ++nt) {
      f16x16 bb = frag_b(buf0, CI, nt * 16, ks * 32, lane);
      acc[nt] = wmma16(a, bb, acc[nt]);
    }
  }
#pragma unroll
  for (int nt = 0; nt < 8; ++nt)
#pragma unroll
    for (int r = 0; r < 8; ++r)
      buf1[(size_t)(strip + hi * 8 + r) * CI + nt * 16 + l] = (_Float16)acc[nt][r];
  __syncthreads();

  // mm2: f = T1 * Theta^T  (A = buf1, B rows = Theta rows indexed by d)
#pragma unroll
  for (int i = 0; i < 8; ++i) acc[i] = z8;
#pragma unroll
  for (int ks = 0; ks < 4; ++ks) {
    f16x16 a = frag_a(buf1, CI, strip, ks * 32, lane);
#pragma unroll
    for (int nt = 0; nt < 8; ++nt) {
      f16x16 bb = frag_b(thetaH, CI, nt * 16, ks * 32, lane);
      acc[nt] = wmma16(a, bb, acc[nt]);
    }
  }
  __syncthreads();  // all reads of buf1 done; it becomes F

  // rank-1 bias terms + in-register row softmax (each row lives in one 16-lane half)
#pragma unroll
  for (int r = 0; r < 8; ++r) {
    int M = strip + hi * 8 + r;
    float pm = pb[M], fm = phs[M];
    float mx = -3.0e38f;
#pragma unroll
    for (int nt = 0; nt < 8; ++nt) {
      int N = nt * 16 + l;
      float v = acc[nt][r] + fm * tb[N] + pm * ths[N] + 4096.f * pm * tb[N];
      acc[nt][r] = v;
      mx = fmaxf(mx, v);
    }
    mx = fmaxf(mx, __shfl_xor(mx, 1, 16));
    mx = fmaxf(mx, __shfl_xor(mx, 2, 16));
    mx = fmaxf(mx, __shfl_xor(mx, 4, 16));
    mx = fmaxf(mx, __shfl_xor(mx, 8, 16));
    float sm = 0.f;
#pragma unroll
    for (int nt = 0; nt < 8; ++nt) {
      float e = __expf(acc[nt][r] - mx);
      acc[nt][r] = e;
      sm += e;
    }
    sm += __shfl_xor(sm, 1, 16);
    sm += __shfl_xor(sm, 2, 16);
    sm += __shfl_xor(sm, 4, 16);
    sm += __shfl_xor(sm, 8, 16);
    float inv = 1.f / sm;
#pragma unroll
    for (int nt = 0; nt < 8; ++nt)
      buf1[(size_t)M * CI + nt * 16 + l] = (_Float16)(acc[nt][r] * inv);
  }
  __syncthreads();

  // mm3: T2 = F * g_w  (B from gwTH = g_w^T rows); store T2 transposed into buf0
#pragma unroll
  for (int i = 0; i < 8; ++i) acc[i] = z8;
#pragma unroll
  for (int ks = 0; ks < 4; ++ks) {
    f16x16 a = frag_a(buf1, CI, strip, ks * 32, lane);
#pragma unroll
    for (int nt = 0; nt < 8; ++nt) {
      f16x16 bb = frag_b(gwTH, CI, nt * 16, ks * 32, lane);
      acc[nt] = wmma16(a, bb, acc[nt]);
    }
  }
#pragma unroll
  for (int nt = 0; nt < 8; ++nt)
#pragma unroll
    for (int r = 0; r < 8; ++r)
      buf0[(size_t)(nt * 16 + l) * CI + (strip + hi * 8 + r)] = (_Float16)acc[nt][r];
  __syncthreads();

  // mm4: A = W * T2 (B rows from buf0 = T2^T); fold BN scale + identity; emit f16
#pragma unroll
  for (int i = 0; i < 8; ++i) acc[i] = z8;
#pragma unroll
  for (int ks = 0; ks < 4; ++ks) {
    f16x16 a = frag_a(WwH, CI, strip, ks * 32, lane);
#pragma unroll
    for (int nt = 0; nt < 8; ++nt) {
      f16x16 bb = frag_b(buf0, CI, nt * 16, ks * 32, lane);
      acc[nt] = wmma16(a, bb, acc[nt]);
    }
  }
  _Float16* Aout = Ah + (size_t)b * CI * CI;
#pragma unroll
  for (int nt = 0; nt < 8; ++nt)
#pragma unroll
    for (int r = 0; r < 8; ++r) {
      int o = strip + hi * 8 + r, ii = nt * 16 + l;
      float v = bsc[o] * acc[nt][r] + ((o == ii) ? 1.f : 0.f);
      Aout[(size_t)o * CI + ii] = (_Float16)v;
    }
  __syncthreads();

  // c' = bnscale*(W*(F g_b) + W_b) + bnshift
  if (t < CI) {
    float a = 0.f;
    for (int d = 0; d < CI; ++d) a += (float)buf1[(size_t)t * CI + d] * g_b[d];
    uv[t] = a;
  }
  __syncthreads();
  if (t < CI) {
    float a = 0.f; const float* r = W_w + (size_t)t * CI;
    for (int c = 0; c < CI; ++c) a += r[c] * uv[c];
    cv[(size_t)b * CI + t] = bsc[t] * (a + W_b[t]) + bsh[t];
  }
}

// ---------------- kernel 3: out = A'' * X + c' (BN + residual already folded) ----------------
// Same software pipeline: prefetch next 64-column chunk into registers,
// WMMA on current LDS buffer, then convert+store, one barrier per chunk.
__global__ __launch_bounds__(256) void k_apply(const float* __restrict__ x,
                                               const _Float16* __restrict__ Ah,
                                               const float* __restrict__ cv,
                                               float* __restrict__ out) {
  __shared__ _Float16 Al[CI * CI];         // 32 KB
  __shared__ _Float16 Xt[2][64 * CI];      // 2 x 16 KB, transposed [n][c]
  __shared__ float cvl[CI];
  const int t = threadIdx.x, wg = blockIdx.x;
  const int b = wg >> 3, ntile = wg & 7;   // 8 n-tiles of 512
  const int lane = t & 31, wave = t >> 5, l = lane & 15, hi = lane >> 4;
  const int strip = wave * 16;
  const _Float16* Ab = Ah + (size_t)b * CI * CI;
#pragma unroll
  for (int i = 0; i < 8; ++i)
    ((int4*)Al)[t + 256 * i] = ((const int4*)Ab)[t + 256 * i];
  if (t < CI) cvl[t] = cv[(size_t)b * CI + t];
  const float* xb = x + (size_t)b * CI * NTOT + ntile * 512;
  float* ob = out + (size_t)b * CI * NTOT + ntile * 512;
  const f32x8 z8 = {0.f, 0.f, 0.f, 0.f, 0.f, 0.f, 0.f, 0.f};

  // prologue: chunk 0 -> buffer 0
  {
    float4 st[8];
#pragma unroll
    for (int i = 0; i < 8; ++i) {
      int idx = t + 256 * i;                 // 2048 float4 = 128 rows x 16 float4
      int row = idx >> 4, c4 = idx & 15;
      st[i] = ((const float4*)(xb + (size_t)row * NTOT))[c4];
    }
#pragma unroll
    for (int i = 0; i < 8; ++i) {
      int idx = t + 256 * i;
      int row = idx >> 4, c4 = idx & 15;
      int n0 = c4 * 4;
      Xt[0][(size_t)(n0 + 0) * CI + row] = (_Float16)st[i].x;
      Xt[0][(size_t)(n0 + 1) * CI + row] = (_Float16)st[i].y;
      Xt[0][(size_t)(n0 + 2) * CI + row] = (_Float16)st[i].z;
      Xt[0][(size_t)(n0 + 3) * CI + row] = (_Float16)st[i].w;
    }
  }
  __syncthreads();

  for (int ch = 0; ch < 8; ++ch) {           // 8 chunks of 64 columns
    const int cur = ch & 1, nxt = cur ^ 1;
    float4 st[8];
    if (ch + 1 < 8) {
#pragma unroll
      for (int i = 0; i < 8; ++i) {
        int idx = t + 256 * i;
        int row = idx >> 4, c4 = idx & 15;
        st[i] = ((const float4*)(xb + (size_t)row * NTOT + (size_t)(ch + 1) * 64))[c4];
      }
    }
    f32x8 acc[4];
#pragma unroll
    for (int i = 0; i < 4; ++i) acc[i] = z8;
#pragma unroll
    for (int ks = 0; ks < 4; ++ks) {
      f16x16 a = frag_a(Al, CI, strip, ks * 32, lane);
#pragma unroll
      for (int nt = 0; nt < 4; ++nt) {
        f16x16 bb = frag_b(&Xt[cur][0], CI, nt * 16, ks * 32, lane);
        acc[nt] = wmma16(a, bb, acc[nt]);
      }
    }
    if (ch + 1 < 8) {
#pragma unroll
      for (int i = 0; i < 8; ++i) {
        int idx = t + 256 * i;
        int row = idx >> 4, c4 = idx & 15;
        int n0 = c4 * 4;
        Xt[nxt][(size_t)(n0 + 0) * CI + row] = (_Float16)st[i].x;
        Xt[nxt][(size_t)(n0 + 1) * CI + row] = (_Float16)st[i].y;
        Xt[nxt][(size_t)(n0 + 2) * CI + row] = (_Float16)st[i].z;
        Xt[nxt][(size_t)(n0 + 3) * CI + row] = (_Float16)st[i].w;
      }
    }
#pragma unroll
    for (int nt = 0; nt < 4; ++nt)
#pragma unroll
      for (int r = 0; r < 8; ++r) {
        int o = strip + hi * 8 + r;
        int n = ch * 64 + nt * 16 + l;
        ob[(size_t)o * NTOT + n] = acc[nt][r] + cvl[o];
      }
    __syncthreads();
  }
}

// ---------------- host launcher ----------------
extern "C" void kernel_launch(void* const* d_in, const int* in_sizes, int n_in,
                              void* d_out, int out_size, void* d_ws, size_t ws_size,
                              hipStream_t stream) {
  (void)in_sizes; (void)n_in; (void)out_size; (void)ws_size;
  const float* x        = (const float*)d_in[0];
  const float* theta_w  = (const float*)d_in[1];
  const float* theta_b  = (const float*)d_in[2];
  const float* phi_w    = (const float*)d_in[3];
  const float* phi_b    = (const float*)d_in[4];
  const float* g_w      = (const float*)d_in[5];
  const float* g_b      = (const float*)d_in[6];
  const float* W_w      = (const float*)d_in[7];
  const float* W_b      = (const float*)d_in[8];
  const float* bn_gamma = (const float*)d_in[9];
  const float* bn_beta  = (const float*)d_in[10];
  const float* bn_mean  = (const float*)d_in[11];
  const float* bn_var   = (const float*)d_in[12];
  float* out = (float*)d_out;

  char* ws = (char*)d_ws;
  constexpr size_t SZ_GP  = (size_t)NBLK * NSPLIT * CI * CI * sizeof(float); // 16 MB
  constexpr size_t SZ_SP  = (size_t)NBLK * NSPLIT * CI * sizeof(float);
  constexpr size_t SZ_W16 = (size_t)CI * CI * sizeof(_Float16);
  constexpr size_t SZ_AH  = (size_t)NBLK * CI * CI * sizeof(_Float16);
  constexpr size_t OFF_GP  = 0;
  constexpr size_t OFF_SP  = OFF_GP + SZ_GP;
  constexpr size_t OFF_PHI = OFF_SP + SZ_SP;
  constexpr size_t OFF_THE = OFF_PHI + SZ_W16;
  constexpr size_t OFF_GWT = OFF_THE + SZ_W16;
  constexpr size_t OFF_WW  = OFF_GWT + SZ_W16;
  constexpr size_t OFF_AH  = OFF_WW + SZ_W16;
  constexpr size_t OFF_CV  = OFF_AH + SZ_AH;

  float*     Gp     = (float*)(ws + OFF_GP);
  float*     sp     = (float*)(ws + OFF_SP);
  _Float16*  phiH   = (_Float16*)(ws + OFF_PHI);
  _Float16*  thetaH = (_Float16*)(ws + OFF_THE);
  _Float16*  gwTH   = (_Float16*)(ws + OFF_GWT);
  _Float16*  WwH    = (_Float16*)(ws + OFF_WW);
  _Float16*  Ah     = (_Float16*)(ws + OFF_AH);
  float*     cvp    = (float*)(ws + OFF_CV);

  k_wconv<<<64, 256, 0, stream>>>(theta_w, phi_w, g_w, W_w, phiH, thetaH, gwTH, WwH);
  k_gram<<<NBLK * NSPLIT, 256, 0, stream>>>(x, Gp, sp);
  k_attn<<<NBLK, 256, 0, stream>>>(Gp, sp, phiH, thetaH, gwTH, WwH,
                                   theta_b, phi_b, g_b, W_w, W_b,
                                   bn_gamma, bn_beta, bn_mean, bn_var,
                                   phi_w, theta_w, Ah, cvp);
  k_apply<<<NBLK * 8, 256, 0, stream>>>(x, Ah, cvp, out);
}
